// SimpleFloodTGCN_86062554677667
// MI455X (gfx1250) — matrix operand order repaired and checked by reading, not compiled, not run on hardware
//
#include <hip/hip_runtime.h>
#include <hip/hip_bf16.h>
#include <math.h>

typedef __attribute__((ext_vector_type(16))) __bf16 v16bf;
typedef __attribute__((ext_vector_type(8)))  float  v8f;
typedef unsigned short ushort_t;

#define NNODES 50000
#define TSTEPS 12
#define INF_   8
#define SF_    16
#define HDIM   128
#define H2DIM  64
#define NEDGE  400000
#define N1D_   10000
#define MTILES 3125   // 50000/16

union ABu { uint4 u[2]; v16bf v; };

__device__ __forceinline__ ushort_t f2bf(float f) {
    unsigned int x = __float_as_uint(f);
    unsigned int r = x + 0x7fffu + ((x >> 16) & 1u);
    return (ushort_t)(r >> 16);
}
__device__ __forceinline__ float gelu_f(float x) {
    return 0.5f * x * (1.0f + erff(x * 0.70710678118654752f));
}
__device__ __forceinline__ float sig_f(float x) {
    return 1.0f / (1.0f + __expf(-x));
}

// ---------------- weight transpose + bf16 convert: W[K,Nout] -> Wt[Nout,K] ----
__global__ void wt_bf16_k(const float* __restrict__ W, ushort_t* __restrict__ Wt,
                          int K, int Nout) {
    int i = blockIdx.x * blockDim.x + threadIdx.x;
    if (i < K * Nout) {
        int k = i / Nout, n = i - k * Nout;
        Wt[(size_t)n * K + k] = f2bf(W[i]);
    }
}

// ---------------- CSR build ------------------------------------------------
__global__ void count_k(const int* __restrict__ er, int* __restrict__ cnt, int E) {
    int e = blockIdx.x * blockDim.x + threadIdx.x;
    if (e < E) atomicAdd(&cnt[er[e]], 1);
}
__global__ void scan_k(const int* __restrict__ cnt, int* __restrict__ start, int n) {
    __shared__ int buf[1024];
    __shared__ int carry;
    int tid = threadIdx.x;
    if (tid == 0) carry = 0;
    __syncthreads();
    for (int base = 0; base < n; base += 1024) {
        int v = (base + tid < n) ? cnt[base + tid] : 0;
        buf[tid] = v;
        __syncthreads();
        for (int off = 1; off < 1024; off <<= 1) {
            int t = (tid >= off) ? buf[tid - off] : 0;
            __syncthreads();
            buf[tid] += t;
            __syncthreads();
        }
        if (base + tid < n) start[base + tid] = carry + buf[tid] - v;
        __syncthreads();
        if (tid == 1023) carry += buf[1023];
        __syncthreads();
    }
    if (tid == 0) start[n] = carry;
}
__global__ void copy_k(const int* __restrict__ a, int* __restrict__ b, int n) {
    int i = blockIdx.x * blockDim.x + threadIdx.x;
    if (i < n) b[i] = a[i];
}
__global__ void scatter_k(const int* __restrict__ er, const int* __restrict__ ec,
                          const float* __restrict__ ev, int* __restrict__ cur,
                          int* __restrict__ ci, float* __restrict__ cv, int E) {
    int e = blockIdx.x * blockDim.x + threadIdx.x;
    if (e < E) {
        int r = er[e];
        int p = atomicAdd(&cur[r], 1);
        ci[p] = ec[e];
        cv[p] = ev[e];
    }
}

// ---- main WMMA GEMM, 32x64 tile per wave: C = A_bf16[M,K](lda) * Wt^T + b ---
__global__ void gemm_wmma32(const ushort_t* __restrict__ A, int lda,
                            const ushort_t* __restrict__ Wt,
                            const float* __restrict__ bias,
                            float* __restrict__ C, int K, int Nout, int mtiles) {
    int wavesPer = blockDim.x >> 5;
    int wid = blockIdx.x * wavesPer + (threadIdx.x >> 5);
    int nblks = Nout >> 6;
    int msup = (mtiles + 1) >> 1;
    if (wid >= msup * nblks) return;
    int ms = wid / nblks, nblk = wid - ms * nblks;
    int m0 = ms * 2, m1 = m0 + 1;
    bool has1 = (m1 < mtiles);
    int lane = threadIdx.x & 31;
    int half = lane >> 4, mr = lane & 15;

    v8f acc[8];
    v8f zero = {};
#pragma unroll
    for (int s = 0; s < 8; s++) acc[s] = zero;

    const ushort_t* arow0 = A + (size_t)(m0 * 16 + mr) * lda;
    const ushort_t* arow1 = A + (size_t)((has1 ? m1 : m0) * 16 + mr) * lda;
    int ncb = nblk * 64;
    for (int kk = 0; kk < K; kk += 32) {
        ABu fa0, fa1;
        fa0.u[0] = *(const uint4*)(arow0 + kk + half * 16);
        fa0.u[1] = *(const uint4*)(arow0 + kk + 8 + half * 16);
        fa1.u[0] = *(const uint4*)(arow1 + kk + half * 16);
        fa1.u[1] = *(const uint4*)(arow1 + kk + 8 + half * 16);
#pragma unroll
        for (int s = 0; s < 4; s++) {
            ABu fb;
            fb.v = *(const v16bf*)(Wt + (size_t)(ncb + s * 16 + mr) * K
                                   + kk + half * 16);
            acc[s]     = __builtin_amdgcn_wmma_f32_16x16x32_bf16(
                false, fa0.v, false, fb.v, (short)0, acc[s], false, false);
            acc[4 + s] = __builtin_amdgcn_wmma_f32_16x16x32_bf16(
                false, fa1.v, false, fb.v, (short)0, acc[4 + s], false, false);
        }
    }
    int rowb0 = m0 * 16 + half * 8;
#pragma unroll
    for (int s = 0; s < 4; s++) {
        int col = ncb + s * 16 + mr;
        float bb = bias[col];
#pragma unroll
        for (int i = 0; i < 8; i++)
            C[(size_t)(rowb0 + i) * Nout + col] = acc[s][i] + bb;
    }
    if (has1) {
        int rowb1 = m1 * 16 + half * 8;
#pragma unroll
        for (int s = 0; s < 4; s++) {
            int col = ncb + s * 16 + mr;
            float bb = bias[col];
#pragma unroll
            for (int i = 0; i < 8; i++)
                C[(size_t)(rowb1 + i) * Nout + col] = acc[4 + s][i] + bb;
        }
    }
}

// ---- head WMMA GEMM (16x64 tile, per-M-tile weight select, GELU epilogue) ---
__global__ void gemm_wmma(const ushort_t* __restrict__ A, int lda,
                          const ushort_t* __restrict__ Wt, const float* __restrict__ bias,
                          const ushort_t* __restrict__ Wt2, const float* __restrict__ bias2,
                          int mtileSplit,
                          float* __restrict__ C, int K, int Nout, int mtiles, int act) {
    int wavesPer = blockDim.x >> 5;
    int wid = blockIdx.x * wavesPer + (threadIdx.x >> 5);
    int nblks = Nout >> 6;
    if (wid >= mtiles * nblks) return;
    int mtile = wid / nblks, nblk = wid - mtile * nblks;
    const ushort_t* Wsel = (mtile < mtileSplit) ? Wt : Wt2;
    const float*    bsel = (mtile < mtileSplit) ? bias : bias2;
    int lane = threadIdx.x & 31;
    int half = lane >> 4, mr = lane & 15;

    v8f acc[4];
    v8f zero = {};
#pragma unroll
    for (int s = 0; s < 4; s++) acc[s] = zero;

    const ushort_t* arow = A + (size_t)(mtile * 16 + mr) * lda;
    int ncolbase = nblk * 64;
    for (int kk = 0; kk < K; kk += 32) {
        ABu fa;
        fa.u[0] = *(const uint4*)(arow + kk + half * 16);
        fa.u[1] = *(const uint4*)(arow + kk + 8 + half * 16);
#pragma unroll
        for (int s = 0; s < 4; s++) {
            ABu fb;
            fb.v = *(const v16bf*)(Wsel + (size_t)(ncolbase + s * 16 + mr) * K
                                   + kk + half * 16);
            acc[s] = __builtin_amdgcn_wmma_f32_16x16x32_bf16(
                false, fa.v, false, fb.v, (short)0, acc[s], false, false);
        }
    }
    int rowb = mtile * 16 + half * 8;
#pragma unroll
    for (int s = 0; s < 4; s++) {
        int col = ncolbase + s * 16 + mr;
        float bb = bsel[col];
#pragma unroll
        for (int i = 0; i < 8; i++) {
            float v = acc[s][i] + bb;
            if (act) v = gelu_f(v);
            C[(size_t)(rowb + i) * Nout + col] = v;
        }
    }
}

// ---------------- SpMM (CSR, wave per row): Y[row] = sum val * X[col] --------
__global__ void spmm_csr(const int* __restrict__ rs, const int* __restrict__ ci,
                         const float* __restrict__ cv, const float* __restrict__ X,
                         float* __restrict__ Y, int C, int Nrows) {
    int row = blockIdx.x * (blockDim.x >> 5) + (threadIdx.x >> 5);
    if (row >= Nrows) return;
    int lane = threadIdx.x & 31;
    int nv = C >> 5;
    float acc[8];
#pragma unroll
    for (int i = 0; i < 8; i++) acc[i] = 0.f;
    int s = rs[row], e = rs[row + 1];
    for (int j = s; j < e; j++) {
        int col = ci[j];
        float v = cv[j];
        const float* xr = X + (size_t)col * C + lane;
#pragma unroll
        for (int i = 0; i < 8; i++)
            if (i < nv) acc[i] += v * xr[i * 32];
    }
    float* yr = Y + (size_t)row * C + lane;
#pragma unroll
    for (int i = 0; i < 8; i++)
        if (i < nv) yr[i * 32] = acc[i];
}

// ---------------- fuse: concat(x_t, static)@Wf + b -> LN -> GELU -> bf16 -----
__global__ void fuse_k(const float* __restrict__ xseq_t, const float* __restrict__ statf,
                       const float* __restrict__ Wf, const float* __restrict__ bf_,
                       const float* __restrict__ g, const float* __restrict__ beta,
                       ushort_t* __restrict__ cat0, ushort_t* __restrict__ cat0r,
                       int Nrows) {
    __shared__ float sW[24 * 64];
    __shared__ float sx[8][24];
    int tid = threadIdx.x;
    for (int i = tid; i < 24 * 64; i += blockDim.x) sW[i] = Wf[i];
    if (tid < 8 * 24) {
        int r = tid / 24, k = tid - r * 24;
        int gr = blockIdx.x * 8 + r;
        if (gr < Nrows)
            sx[r][k] = (k < 8) ? xseq_t[(size_t)gr * 8 + k]
                               : statf[(size_t)gr * 16 + (k - 8)];
    }
    __syncthreads();
    int wid = tid >> 5, lane = tid & 31;
    int row = blockIdx.x * 8 + wid;
    if (row >= Nrows) return;
    float z0 = bf_[lane], z1 = bf_[lane + 32];
#pragma unroll
    for (int k = 0; k < 24; k++) {
        float xv = sx[wid][k];
        z0 += xv * sW[k * 64 + lane];
        z1 += xv * sW[k * 64 + lane + 32];
    }
    float s = z0 + z1;
    for (int m = 16; m; m >>= 1) s += __shfl_xor(s, m);
    float mean = s * (1.f / 64.f);
    float d0 = z0 - mean, d1 = z1 - mean;
    float vs = d0 * d0 + d1 * d1;
    for (int m = 16; m; m >>= 1) vs += __shfl_xor(vs, m);
    float inv = rsqrtf(vs * (1.f / 64.f) + 1e-5f);
    float y0 = gelu_f(d0 * inv * g[lane] + beta[lane]);
    float y1 = gelu_f(d1 * inv * g[lane + 32] + beta[lane + 32]);
    ushort_t b0 = f2bf(y0), b1 = f2bf(y1);
    size_t base = (size_t)row * 192;
    cat0[base + lane] = b0;  cat0[base + lane + 32] = b1;
    cat0r[base + lane] = b0; cat0r[base + lane + 32] = b1;
}

// ---------------- r/u gate: r=sig(g[:, :128]), u=sig(g[:,128:]); catr=bf16(r*h)
__global__ void ru_k(const float* __restrict__ S, const float* __restrict__ h,
                     float* __restrict__ u, ushort_t* __restrict__ catr,
                     int ld, int off, int Nrows) {
    int i = blockIdx.x * blockDim.x + threadIdx.x;
    if (i >= Nrows * 128) return;
    int row = i >> 7, c = i & 127;
    float r  = sig_f(S[(size_t)row * 256 + c]);
    float uu = sig_f(S[(size_t)row * 256 + 128 + c]);
    u[i] = uu;
    catr[(size_t)row * ld + off + c] = f2bf(r * h[i]);
}

// ---------------- h update + LN (wave per row), fan-out bf16 copies ----------
__global__ void hupdate_ln(const float* __restrict__ cpre, const float* __restrict__ u,
                           float* __restrict__ h, const float* __restrict__ lng,
                           const float* __restrict__ lnb,
                           ushort_t* d1, int ld1, int off1,
                           ushort_t* d2, int ld2, int off2,
                           ushort_t* d3, int ld3, int off3, int Nrows) {
    int row = blockIdx.x * (blockDim.x >> 5) + (threadIdx.x >> 5);
    if (row >= Nrows) return;
    int lane = threadIdx.x & 31;
    size_t rb = (size_t)row * 128;
    float hn[4];
    float s = 0.f;
#pragma unroll
    for (int i = 0; i < 4; i++) {
        int c = lane + 32 * i;
        float uu = u[rb + c];
        float cc = tanhf(cpre[rb + c]);
        float hv = h[rb + c];
        hn[i] = uu * hv + (1.f - uu) * cc;
        s += hn[i];
    }
    for (int m = 16; m; m >>= 1) s += __shfl_xor(s, m);
    float mean = s * (1.f / 128.f);
    float vs = 0.f;
#pragma unroll
    for (int i = 0; i < 4; i++) { hn[i] -= mean; vs += hn[i] * hn[i]; }
    for (int m = 16; m; m >>= 1) vs += __shfl_xor(vs, m);
    float inv = rsqrtf(vs * (1.f / 128.f) + 1e-5f);
#pragma unroll
    for (int i = 0; i < 4; i++) {
        int c = lane + 32 * i;
        float y = hn[i] * inv * lng[c] + lnb[c];
        h[rb + c] = y;
        ushort_t bb = f2bf(y);
        if (d1) d1[(size_t)row * ld1 + off1 + c] = bb;
        if (d2) d2[(size_t)row * ld2 + off2 + c] = bb;
        if (d3) d3[(size_t)row * ld3 + off3 + c] = bb;
    }
}

// ---------------- head stage 2: out = depth + scale*(G@W2 + b2) --------------
__global__ void head2_k(const float* __restrict__ G,
                        const float* __restrict__ w2a, const float* __restrict__ b2a,
                        const float* __restrict__ w2b, const float* __restrict__ b2b,
                        const float* __restrict__ depth, const float* __restrict__ scale,
                        float* __restrict__ out, int Nrows, int n1d) {
    int row = blockIdx.x * (blockDim.x >> 5) + (threadIdx.x >> 5);
    if (row >= Nrows) return;
    int lane = threadIdx.x & 31;
    const float* w2 = (row < n1d) ? w2a : w2b;
    float b2 = (row < n1d) ? b2a[0] : b2b[0];
    const float* g = G + (size_t)row * 64;
    float s = g[lane] * w2[lane] + g[lane + 32] * w2[lane + 32];
    for (int m = 16; m; m >>= 1) s += __shfl_xor(s, m);
    if (lane == 0) out[row] = depth[row] + scale[0] * (s + b2);
}

// =============================================================================
extern "C" void kernel_launch(void* const* d_in, const int* in_sizes, int n_in,
                              void* d_out, int out_size, void* d_ws, size_t ws_size,
                              hipStream_t stream) {
    (void)in_sizes; (void)n_in; (void)out_size; (void)ws_size;
    const float* node_seq  = (const float*)d_in[0];
    const float* depth     = (const float*)d_in[1];
    const float* statf     = (const float*)d_in[2];
    const int*   erow      = (const int*)d_in[3];
    const int*   ecol      = (const int*)d_in[4];
    const float* eval_     = (const float*)d_in[5];
    const float* fuse_W    = (const float*)d_in[6];
    const float* fuse_b    = (const float*)d_in[7];
    const float* fuse_g    = (const float*)d_in[8];
    const float* fuse_beta = (const float*)d_in[9];
    const float* Wg0 = (const float*)d_in[10]; const float* bg0 = (const float*)d_in[11];
    const float* Wc0 = (const float*)d_in[12]; const float* bc0 = (const float*)d_in[13];
    const float* lng0= (const float*)d_in[14]; const float* lnb0= (const float*)d_in[15];
    const float* Wg1 = (const float*)d_in[16]; const float* bg1 = (const float*)d_in[17];
    const float* Wc1 = (const float*)d_in[18]; const float* bc1 = (const float*)d_in[19];
    const float* lng1= (const float*)d_in[20]; const float* lnb1= (const float*)d_in[21];
    const float* h1W1= (const float*)d_in[22]; const float* h1b1= (const float*)d_in[23];
    const float* h1W2= (const float*)d_in[24]; const float* h1b2= (const float*)d_in[25];
    const float* h2W1= (const float*)d_in[26]; const float* h2b1= (const float*)d_in[27];
    const float* h2W2= (const float*)d_in[28]; const float* h2b2= (const float*)d_in[29];
    const float* res_scale = (const float*)d_in[30];
    float* out = (float*)d_out;

    // ---- workspace bump allocator (256B aligned) ----
    size_t off = 0;
    auto alloc = [&](size_t bytes) -> void* {
        off = (off + 255) & ~(size_t)255;
        void* p = (char*)d_ws + off;
        off += bytes;
        return p;
    };
    const int N = NNODES, E = NEDGE;
    ushort_t* wg0t  = (ushort_t*)alloc(256 * 192 * 2);
    ushort_t* wc0t  = (ushort_t*)alloc(128 * 192 * 2);
    ushort_t* wg1t  = (ushort_t*)alloc(256 * 256 * 2);
    ushort_t* wc1t  = (ushort_t*)alloc(128 * 256 * 2);
    ushort_t* w1at  = (ushort_t*)alloc(64 * 128 * 2);
    ushort_t* w1bt  = (ushort_t*)alloc(64 * 128 * 2);
    ushort_t* cat0  = (ushort_t*)alloc((size_t)N * 192 * 2);
    ushort_t* cat0r = (ushort_t*)alloc((size_t)N * 192 * 2);
    ushort_t* cat1  = (ushort_t*)alloc((size_t)N * 256 * 2);
    ushort_t* cat1r = (ushort_t*)alloc((size_t)N * 256 * 2);
    float* Z  = (float*)alloc((size_t)N * 256 * 4);
    float* Sb = (float*)alloc((size_t)N * 256 * 4);
    float* u  = (float*)alloc((size_t)N * 128 * 4);
    float* h0 = (float*)alloc((size_t)N * 128 * 4);
    float* h1 = (float*)alloc((size_t)N * 128 * 4);
    float* G  = (float*)alloc((size_t)N * 64 * 4);
    int*   cnt    = (int*)alloc((size_t)(N + 1) * 4);
    int*   rstart = (int*)alloc((size_t)(N + 1) * 4);
    int*   cursor = (int*)alloc((size_t)(N + 1) * 4);
    int*   ci     = (int*)alloc((size_t)E * 4);
    float* cv     = (float*)alloc((size_t)E * 4);

    // ---- init (h=0, concat h-halves=0 for step 0) ----
    hipMemsetAsync(h0, 0, (size_t)N * 128 * 4, stream);
    hipMemsetAsync(h1, 0, (size_t)N * 128 * 4, stream);
    hipMemsetAsync(cat0, 0, (size_t)N * 192 * 2, stream);
    hipMemsetAsync(cat1, 0, (size_t)N * 256 * 2, stream);
    hipMemsetAsync(cnt, 0, (size_t)(N + 1) * 4, stream);

    // ---- bf16 transposed weights ----
    auto wt = [&](const float* W, ushort_t* Wt, int K, int Nout) {
        int tot = K * Nout;
        wt_bf16_k<<<(tot + 255) / 256, 256, 0, stream>>>(W, Wt, K, Nout);
    };
    wt(Wg0, wg0t, 192, 256);
    wt(Wc0, wc0t, 192, 128);
    wt(Wg1, wg1t, 256, 256);
    wt(Wc1, wc1t, 256, 128);
    wt(h1W1, w1at, 128, 64);
    wt(h2W1, w1bt, 128, 64);

    // ---- CSR build ----
    count_k<<<(E + 255) / 256, 256, 0, stream>>>(erow, cnt, E);
    scan_k<<<1, 1024, 0, stream>>>(cnt, rstart, N);
    copy_k<<<(N + 256) / 256, 256, 0, stream>>>(rstart, cursor, N + 1);
    scatter_k<<<(E + 255) / 256, 256, 0, stream>>>(erow, ecol, eval_, cursor, ci, cv, E);

    // big GEMMs: 32x64 tile per wave
    auto launch_gemm32 = [&](const ushort_t* A, int lda, const ushort_t* Wt,
                             const float* bias, float* Cp, int K, int Nout) {
        int nblks = Nout >> 6;
        int total = ((MTILES + 1) >> 1) * nblks;
        gemm_wmma32<<<(total + 7) / 8, 256, 0, stream>>>(A, lda, Wt, bias,
                                                         Cp, K, Nout, MTILES);
    };
    const int rowBlocks = (N + 7) / 8;   // wave-per-row kernels, 8 waves/block

    // ---- time loop ----
    for (int t = 0; t < TSTEPS; t++) {
        const float* xseq_t = node_seq + (size_t)t * N * INF_;
        fuse_k<<<rowBlocks, 256, 0, stream>>>(xseq_t, statf, fuse_W, fuse_b,
                                              fuse_g, fuse_beta, cat0, cat0r, N);
        // ---- layer 0 (K = 64+128 = 192) ----
        launch_gemm32(cat0, 192, wg0t, bg0, Z, 192, 256);
        spmm_csr<<<rowBlocks, 256, 0, stream>>>(rstart, ci, cv, Z, Sb, 256, N);
        ru_k<<<(N * 128 + 255) / 256, 256, 0, stream>>>(Sb, h0, u, cat0r, 192, 64, N);
        launch_gemm32(cat0r, 192, wc0t, bc0, Z, 192, 128);
        spmm_csr<<<rowBlocks, 256, 0, stream>>>(rstart, ci, cv, Z, Sb, 128, N);
        hupdate_ln<<<rowBlocks, 256, 0, stream>>>(Sb, u, h0, lng0, lnb0,
                                                  cat1, 256, 0, cat1r, 256, 0,
                                                  cat0, 192, 64, N);
        // ---- layer 1 (K = 128+128 = 256), x_t = h0 ----
        launch_gemm32(cat1, 256, wg1t, bg1, Z, 256, 256);
        spmm_csr<<<rowBlocks, 256, 0, stream>>>(rstart, ci, cv, Z, Sb, 256, N);
        ru_k<<<(N * 128 + 255) / 256, 256, 0, stream>>>(Sb, h1, u, cat1r, 256, 128, N);
        launch_gemm32(cat1r, 256, wc1t, bc1, Z, 256, 128);
        spmm_csr<<<rowBlocks, 256, 0, stream>>>(rstart, ci, cv, Z, Sb, 128, N);
        hupdate_ln<<<rowBlocks, 256, 0, stream>>>(Sb, u, h1, lng1, lnb1,
                                                  cat1, 256, 128,
                                                  (ushort_t*)nullptr, 0, 0,
                                                  (ushort_t*)nullptr, 0, 0, N);
    }

    // ---- heads: bf16(h1) lives in cat1[:,128:256] (lda=256); split at tile 625
    {
        int total = MTILES * 1;  // Nout=64 -> 1 nblk
        gemm_wmma<<<(total + 7) / 8, 256, 0, stream>>>(
            cat1 + 128, 256, w1at, h1b1, w1bt, h2b1, N1D_ / 16,
            G, 128, 64, MTILES, 1);
    }
    head2_k<<<rowBlocks, 256, 0, stream>>>(G, h1W2, h1b2, h2W2, h2b2,
                                           depth, res_scale, out, N, N1D_);
}
